// GraphEncoder_85237920956533
// MI455X (gfx1250) — compile-verified, matrix operand order checked by reading, not run on hardware
//
#include <hip/hip_runtime.h>
#include <hip/hip_bf16.h>

// ---------------- problem constants (compile-time) ----------------
constexpr int B_  = 4;
constexpr int T_  = 1024;
constexpr int F_  = 128;
constexpr int D_  = 512;
constexpr int H_  = 8;
constexpr int L_  = 4;
constexpr int DFF_ = 2048;
constexpr int HD_ = 64;
constexpr int NE_ = 16;
constexpr int MD_ = 20;
constexpr int BT_ = B_ * T_;          // 4096 rows
constexpr float SCALE_ = 0.125f;      // 1/sqrt(64)

typedef __attribute__((ext_vector_type(16))) _Float16 v16h;
typedef __attribute__((ext_vector_type(8)))  _Float16 v8h;
typedef __attribute__((ext_vector_type(8)))  float    v8f;

// Load a 16-half WMMA fragment (A or B operand) for this lane.
// Per-lane layout of 16-bit operands (ISA 7.12.2): half e holds
// k = (e/8)*16 + hi*8 + (e%8), i.e. two contiguous 8-half (16B) runs.
// `base` must already point at rowStart + hi*8.
__device__ __forceinline__ v16h load_frag(const _Float16* __restrict__ base) {
    v8h lo = *reinterpret_cast<const v8h*>(base);
    v8h hi = *reinterpret_cast<const v8h*>(base + 16);
    return __builtin_shufflevector(lo, hi, 0,1,2,3,4,5,6,7,8,9,10,11,12,13,14,15);
}

// ---------------- generic WMMA GEMM, 32x64 macro-tile per wave ----------------
// C[M,N] = A[M,K] (f16, row-major) x Bt[N,K] (f16, "transposed" weight) + bias
// Each wave: 2 (M) x 4 (N) accumulator tiles -> 8 WMMAs per 32-wide k-step,
// 4x register reuse of A frags, 2x of B frags.
// MODE: 0 = outF32 += val (residual)        1 = outF16 = val
//       2 = outF16 = silu(val)              3 = Q/K scatter [B,H,T,HD] f16
//       4 = V scatter transposed [B,H,HD,T] 5 = outF32 = val
template<int M, int N, int K, int MODE>
__global__ __launch_bounds__(256) void gemm_wmma(
    const _Float16* __restrict__ A,
    const _Float16* __restrict__ Bt,
    const float*    __restrict__ bias,
    float*          __restrict__ outF32,
    _Float16*       __restrict__ outF16)
{
    const int wave = threadIdx.x >> 5;
    const int lane = threadIdx.x & 31;
    const int tile = blockIdx.x * 8 + wave;          // exact multiple: no bounds check
    constexpr int TN = N / 64;                       // macro-tiles across N
    const int m0 = (tile / TN) * 32;
    const int n0 = (tile % TN) * 64;

    const int mr = lane & 15;       // row within a 16-row tile for this lane
    const int hi = lane >> 4;       // k-group select

    const _Float16* arow0 = A  + (size_t)(m0 +  0 + mr) * K + hi * 8;
    const _Float16* arow1 = A  + (size_t)(m0 + 16 + mr) * K + hi * 8;
    const _Float16* brow0 = Bt + (size_t)(n0 +  0 + mr) * K + hi * 8;
    const _Float16* brow1 = Bt + (size_t)(n0 + 16 + mr) * K + hi * 8;
    const _Float16* brow2 = Bt + (size_t)(n0 + 32 + mr) * K + hi * 8;
    const _Float16* brow3 = Bt + (size_t)(n0 + 48 + mr) * K + hi * 8;

    v8f acc[2][4];
#pragma unroll
    for (int mt = 0; mt < 2; ++mt)
#pragma unroll
        for (int nt = 0; nt < 4; ++nt) acc[mt][nt] = {};

    for (int k0 = 0; k0 < K; k0 += 32) {
        v16h a0 = load_frag(arow0 + k0);
        v16h a1 = load_frag(arow1 + k0);
        v16h b0 = load_frag(brow0 + k0);
        v16h b1 = load_frag(brow1 + k0);
        v16h b2 = load_frag(brow2 + k0);
        v16h b3 = load_frag(brow3 + k0);
        // prefetch next k-panel (lowers to global_prefetch_b8; operands live in
        // the workspace so the tail prefetch stays in mapped memory)
        __builtin_prefetch(arow0 + k0 + 32, 0, 3);
        __builtin_prefetch(brow0 + k0 + 32, 0, 3);
        acc[0][0] = __builtin_amdgcn_wmma_f32_16x16x32_f16(false, a0, false, b0, (short)0, acc[0][0], false, false);
        acc[0][1] = __builtin_amdgcn_wmma_f32_16x16x32_f16(false, a0, false, b1, (short)0, acc[0][1], false, false);
        acc[0][2] = __builtin_amdgcn_wmma_f32_16x16x32_f16(false, a0, false, b2, (short)0, acc[0][2], false, false);
        acc[0][3] = __builtin_amdgcn_wmma_f32_16x16x32_f16(false, a0, false, b3, (short)0, acc[0][3], false, false);
        acc[1][0] = __builtin_amdgcn_wmma_f32_16x16x32_f16(false, a1, false, b0, (short)0, acc[1][0], false, false);
        acc[1][1] = __builtin_amdgcn_wmma_f32_16x16x32_f16(false, a1, false, b1, (short)0, acc[1][1], false, false);
        acc[1][2] = __builtin_amdgcn_wmma_f32_16x16x32_f16(false, a1, false, b2, (short)0, acc[1][2], false, false);
        acc[1][3] = __builtin_amdgcn_wmma_f32_16x16x32_f16(false, a1, false, b3, (short)0, acc[1][3], false, false);
    }

    const int nrl = lane & 15;
#pragma unroll
    for (int mt = 0; mt < 2; ++mt) {
#pragma unroll
        for (int nt = 0; nt < 4; ++nt) {
            const int n  = n0 + nt * 16 + nrl;
            const float bv = bias[n];
#pragma unroll
            for (int r = 0; r < 8; ++r) {
                const int mm = m0 + mt * 16 + r + hi * 8; // C layout: VGPR r -> M = r + hi*8
                float val = acc[mt][nt][r] + bv;
                if constexpr (MODE == 0) {
                    outF32[(size_t)mm * N + n] += val;
                } else if constexpr (MODE == 5) {
                    outF32[(size_t)mm * N + n] = val;
                } else if constexpr (MODE == 1) {
                    outF16[(size_t)mm * N + n] = (_Float16)val;
                } else if constexpr (MODE == 2) {
                    float sg = 1.0f / (1.0f + __expf(-val));
                    outF16[(size_t)mm * N + n] = (_Float16)(val * sg);
                } else if constexpr (MODE == 3) {            // [B*T, D] -> [B,H,T,HD]
                    int b = mm >> 10, t = mm & 1023, hh = n >> 6, d = n & 63;
                    outF16[(((size_t)(b * H_ + hh) * T_ + t) * HD_) + d] = (_Float16)val;
                } else {                                     // MODE 4: [B*T, D] -> [B,H,HD,T]
                    int b = mm >> 10, t = mm & 1023, hh = n >> 6, d = n & 63;
                    outF16[(((size_t)(b * H_ + hh) * HD_ + d) * T_) + t] = (_Float16)val;
                }
            }
        }
    }
}

// ---------------- attention: per-wave (b, head, 16-row i-tile) ----------------
// Pass 1: S = Q K^T * scale + graph bias  -> LDS (16 x T f32)
// softmax rows in LDS (wave32 shuffle reductions)
// Pass 2: O = P V   (P read from LDS, cvt to f16)
__global__ __launch_bounds__(64) void attn_kernel(
    const _Float16* __restrict__ q,    // [B,H,T,HD] f16
    const _Float16* __restrict__ k,    // [B,H,T,HD] f16
    const _Float16* __restrict__ vT,   // [B,H,HD,T] f16
    const int*      __restrict__ edge_types,  // [B,T,T]
    const int*      __restrict__ sp_dist,     // [B,T,T]
    const float*    __restrict__ edge_emb,    // [NE,H]
    const float*    __restrict__ dist_emb,    // [MD+1,H]
    _Float16*       __restrict__ o)    // [B,T,H,HD] f16  (== [B*T, D])
{
    extern __shared__ float smem[];                  // 2 waves * 16 * T floats
    const int wave = threadIdx.x >> 5;
    const int lane = threadIdx.x & 31;
    float* S = smem + (size_t)wave * 16 * T_;

    const int gidx = blockIdx.x * 2 + wave;          // over B*H*(T/16) = 2048
    const int it = gidx % (T_ / 16);
    const int bh = gidx / (T_ / 16);
    const int hh = bh % H_;
    const int bb = bh / H_;
    const int i0 = it * 16;

    const _Float16* Q  = q  + (size_t)bh * T_ * HD_;
    const _Float16* Kp = k  + (size_t)bh * T_ * HD_;
    const _Float16* Vt = vT + (size_t)bh * HD_ * T_;

    const int mr = lane & 15;
    const int hi = lane >> 4;

    // Q fragments for the two k-steps of HD=64
    v16h qa0 = load_frag(Q + (size_t)(i0 + mr) * HD_ + 0  + hi * 8);
    v16h qa1 = load_frag(Q + (size_t)(i0 + mr) * HD_ + 32 + hi * 8);

    // ---- pass 1: scores + bias into LDS ----
    for (int j0 = 0; j0 < T_; j0 += 16) {
        v8f c = {};
        v16h b0 = load_frag(Kp + (size_t)(j0 + mr) * HD_ + 0  + hi * 8);
        v16h b1 = load_frag(Kp + (size_t)(j0 + mr) * HD_ + 32 + hi * 8);
        c = __builtin_amdgcn_wmma_f32_16x16x32_f16(false, qa0, false, b0, (short)0, c, false, false);
        c = __builtin_amdgcn_wmma_f32_16x16x32_f16(false, qa1, false, b1, (short)0, c, false, false);

        const int j = j0 + (lane & 15);
#pragma unroll
        for (int r = 0; r < 8; ++r) {
            const int mm = r + hi * 8;
            const int i  = i0 + mm;
            const size_t eidx = ((size_t)(bb * T_ + i)) * T_ + j;
            int et = edge_types[eidx];
            int sd = sp_dist[eidx];
            sd = sd < 0 ? 0 : (sd > MD_ ? MD_ : sd);
            float bias = edge_emb[et * H_ + hh] + dist_emb[sd * H_ + hh];
            S[mm * T_ + j] = c[r] * SCALE_ + bias;
        }
    }
    __syncthreads();

    // ---- softmax over each of the 16 rows ----
    for (int r = 0; r < 16; ++r) {
        float* row = S + r * T_;
        float mx = -3.4e38f;
        for (int j = lane; j < T_; j += 32) mx = fmaxf(mx, row[j]);
#pragma unroll
        for (int off = 16; off > 0; off >>= 1) mx = fmaxf(mx, __shfl_xor(mx, off, 32));
        float sum = 0.0f;
        for (int j = lane; j < T_; j += 32) {
            float e = __expf(row[j] - mx);
            row[j] = e;
            sum += e;
        }
#pragma unroll
        for (int off = 16; off > 0; off >>= 1) sum += __shfl_xor(sum, off, 32);
        float inv = 1.0f / sum;
        for (int j = lane; j < T_; j += 32) row[j] *= inv;
    }
    __syncthreads();

    // ---- pass 2: O = P @ V  (4 output n-tiles of 16 over HD=64) ----
    v8f oc0 = {}, oc1 = {}, oc2 = {}, oc3 = {};
    for (int j0 = 0; j0 < T_; j0 += 32) {
        v16h pa;
        const float* prow = S + mr * T_ + j0 + hi * 8;
#pragma unroll
        for (int e = 0; e < 8; ++e) {
            pa[e]     = (_Float16)prow[e];
            pa[e + 8] = (_Float16)prow[e + 16];
        }
        v16h vb0 = load_frag(Vt + (size_t)( 0 + mr) * T_ + j0 + hi * 8);
        v16h vb1 = load_frag(Vt + (size_t)(16 + mr) * T_ + j0 + hi * 8);
        v16h vb2 = load_frag(Vt + (size_t)(32 + mr) * T_ + j0 + hi * 8);
        v16h vb3 = load_frag(Vt + (size_t)(48 + mr) * T_ + j0 + hi * 8);
        oc0 = __builtin_amdgcn_wmma_f32_16x16x32_f16(false, pa, false, vb0, (short)0, oc0, false, false);
        oc1 = __builtin_amdgcn_wmma_f32_16x16x32_f16(false, pa, false, vb1, (short)0, oc1, false, false);
        oc2 = __builtin_amdgcn_wmma_f32_16x16x32_f16(false, pa, false, vb2, (short)0, oc2, false, false);
        oc3 = __builtin_amdgcn_wmma_f32_16x16x32_f16(false, pa, false, vb3, (short)0, oc3, false, false);
    }

    const int nr = lane & 15;
#pragma unroll
    for (int r = 0; r < 8; ++r) {
        const int i = i0 + r + hi * 8;
        const size_t rowbase = ((size_t)(bb * T_ + i) * H_ + hh) * HD_;
        o[rowbase +  0 + nr] = (_Float16)oc0[r];
        o[rowbase + 16 + nr] = (_Float16)oc1[r];
        o[rowbase + 32 + nr] = (_Float16)oc2[r];
        o[rowbase + 48 + nr] = (_Float16)oc3[r];
    }
}

// ---------------- small helper kernels ----------------
__global__ void transpose_cvt(const float* __restrict__ src, _Float16* __restrict__ dst,
                              int R, int C) {
    int idx = blockIdx.x * 256 + threadIdx.x;
    if (idx >= R * C) return;
    int r = idx / C, c = idx % C;
    dst[(size_t)c * R + r] = (_Float16)src[idx];
}

__global__ void cvt_f32_f16(const float* __restrict__ src, _Float16* __restrict__ dst, int n) {
    int i = blockIdx.x * 256 + threadIdx.x;
    if (i < n) dst[i] = (_Float16)src[i];
}

__global__ void cent_add(float* __restrict__ h, const float* __restrict__ cent,
                         const float* __restrict__ cW, const float* __restrict__ cb) {
    int idx = blockIdx.x * 256 + threadIdx.x;   // BT_*D_ exact
    int bt = idx / D_, d = idx % D_;
    h[idx] += cent[bt] * cW[d] + cb[d];
}

__global__ __launch_bounds__(256) void layernorm(const float* __restrict__ h,
                                                 const float* __restrict__ w,
                                                 const float* __restrict__ b,
                                                 _Float16* __restrict__ x) {
    const int wave = threadIdx.x >> 5;
    const int lane = threadIdx.x & 31;
    const int row = blockIdx.x * 8 + wave;      // BT_ rows, exact
    const float* hr = h + (size_t)row * D_;
    float s = 0.0f;
    for (int i = lane; i < D_; i += 32) s += hr[i];
#pragma unroll
    for (int off = 16; off > 0; off >>= 1) s += __shfl_xor(s, off, 32);
    float mean = s * (1.0f / D_);
    float v = 0.0f;
    for (int i = lane; i < D_; i += 32) { float d = hr[i] - mean; v += d * d; }
#pragma unroll
    for (int off = 16; off > 0; off >>= 1) v += __shfl_xor(v, off, 32);
    float rstd = rsqrtf(v * (1.0f / D_) + 1e-5f);
    for (int i = lane; i < D_; i += 32)
        x[(size_t)row * D_ + i] = (_Float16)((hr[i] - mean) * rstd * w[i] + b[i]);
}

// ---------------- host orchestration ----------------
extern "C" void kernel_launch(void* const* d_in, const int* in_sizes, int n_in,
                              void* d_out, int out_size, void* d_ws, size_t ws_size,
                              hipStream_t stream) {
    const float* node_features = (const float*)d_in[0];
    const float* centrality    = (const float*)d_in[1];
    const int*   edge_types    = (const int*)  d_in[2];
    const int*   sp_dist       = (const int*)  d_in[3];
    const float* node_W        = (const float*)d_in[4];
    const float* node_b        = (const float*)d_in[5];
    const float* cent_W        = (const float*)d_in[6];
    const float* cent_b        = (const float*)d_in[7];
    const float* edge_emb      = (const float*)d_in[8];
    const float* dist_emb      = (const float*)d_in[9];
    const float* ln1_w         = (const float*)d_in[10];
    const float* ln1_b         = (const float*)d_in[11];
    const float* q_W           = (const float*)d_in[12];
    const float* q_b           = (const float*)d_in[13];
    const float* k_W           = (const float*)d_in[14];
    const float* k_b           = (const float*)d_in[15];
    const float* v_W           = (const float*)d_in[16];
    const float* v_b           = (const float*)d_in[17];
    const float* o_W           = (const float*)d_in[18];
    const float* o_b           = (const float*)d_in[19];
    const float* ln2_w         = (const float*)d_in[20];
    const float* ln2_b         = (const float*)d_in[21];
    const float* f1_W          = (const float*)d_in[22];
    const float* f1_b          = (const float*)d_in[23];
    const float* f2_W          = (const float*)d_in[24];
    const float* f2_b          = (const float*)d_in[25];

    // workspace carve-out (~73 MB)
    char* wp = (char*)d_ws;
    auto carve = [&](size_t bytes) -> void* {
        void* p = (void*)wp;
        wp += (bytes + 255) & ~(size_t)255;
        return p;
    };
    float*    h      = (float*)   carve((size_t)BT_ * D_ * 4);
    _Float16* x      = (_Float16*)carve((size_t)BT_ * D_ * 2);
    _Float16* qbuf   = (_Float16*)carve((size_t)BT_ * D_ * 2);
    _Float16* kbuf   = (_Float16*)carve((size_t)BT_ * D_ * 2);
    _Float16* vTbuf  = (_Float16*)carve((size_t)BT_ * D_ * 2);
    _Float16* obuf   = (_Float16*)carve((size_t)BT_ * D_ * 2);
    _Float16* ubuf   = (_Float16*)carve((size_t)BT_ * DFF_ * 2);
    _Float16* nf16   = (_Float16*)carve((size_t)BT_ * F_ * 2);
    _Float16* nodeWt = (_Float16*)carve((size_t)D_ * F_ * 2);
    _Float16* qWt    = (_Float16*)carve((size_t)L_ * D_ * D_ * 2);
    _Float16* kWt    = (_Float16*)carve((size_t)L_ * D_ * D_ * 2);
    _Float16* vWt    = (_Float16*)carve((size_t)L_ * D_ * D_ * 2);
    _Float16* oWt    = (_Float16*)carve((size_t)L_ * D_ * D_ * 2);
    _Float16* f1Wt   = (_Float16*)carve((size_t)L_ * DFF_ * D_ * 2);
    _Float16* f2Wt   = (_Float16*)carve((size_t)L_ * D_ * DFF_ * 2);

    // ---- weight prep: f32 -> f16, transposed to [N,K] ----
    {
        int n = BT_ * F_;
        cvt_f32_f16<<<(n + 255) / 256, 256, 0, stream>>>(node_features, nf16, n);
        int e = F_ * D_;
        transpose_cvt<<<(e + 255) / 256, 256, 0, stream>>>(node_W, nodeWt, F_, D_);
        for (int l = 0; l < L_; ++l) {
            int edd = D_ * D_;
            transpose_cvt<<<(edd + 255) / 256, 256, 0, stream>>>(q_W + (size_t)l * edd, qWt + (size_t)l * edd, D_, D_);
            transpose_cvt<<<(edd + 255) / 256, 256, 0, stream>>>(k_W + (size_t)l * edd, kWt + (size_t)l * edd, D_, D_);
            transpose_cvt<<<(edd + 255) / 256, 256, 0, stream>>>(v_W + (size_t)l * edd, vWt + (size_t)l * edd, D_, D_);
            transpose_cvt<<<(edd + 255) / 256, 256, 0, stream>>>(o_W + (size_t)l * edd, oWt + (size_t)l * edd, D_, D_);
            int ef = D_ * DFF_;
            transpose_cvt<<<(ef + 255) / 256, 256, 0, stream>>>(f1_W + (size_t)l * ef, f1Wt + (size_t)l * ef, D_, DFF_);
            transpose_cvt<<<(ef + 255) / 256, 256, 0, stream>>>(f2_W + (size_t)l * ef, f2Wt + (size_t)l * ef, DFF_, D_);
        }
    }

    // ---- embedding: h = nf @ node_W + node_b ; += centrality * cent_W + cent_b ----
    gemm_wmma<BT_, D_, F_, 5><<<(BT_/32)*(D_/64)/8, 256, 0, stream>>>(nf16, nodeWt, node_b, h, nullptr);
    cent_add<<<BT_ * D_ / 256, 256, 0, stream>>>(h, centrality, cent_W, cent_b);

    // ---- transformer layers ----
    const int gemm_dd_blocks  = (BT_/32) * (D_/64)   / 8;   // 128
    const int gemm_ff1_blocks = (BT_/32) * (DFF_/64) / 8;   // 512
    const int attn_blocks     = B_ * H_ * (T_ / 16) / 2;    // 1024 (2 waves/block)
    const size_t attn_lds     = (size_t)2 * 16 * T_ * sizeof(float); // 128 KB

    for (int l = 0; l < L_; ++l) {
        const size_t dd = (size_t)l * D_ * D_;
        const size_t df = (size_t)l * D_ * DFF_;
        layernorm<<<BT_ / 8, 256, 0, stream>>>(h, ln1_w + l * D_, ln1_b + l * D_, x);
        gemm_wmma<BT_, D_, D_, 3><<<gemm_dd_blocks, 256, 0, stream>>>(x, qWt + dd, q_b + l * D_, nullptr, qbuf);
        gemm_wmma<BT_, D_, D_, 3><<<gemm_dd_blocks, 256, 0, stream>>>(x, kWt + dd, k_b + l * D_, nullptr, kbuf);
        gemm_wmma<BT_, D_, D_, 4><<<gemm_dd_blocks, 256, 0, stream>>>(x, vWt + dd, v_b + l * D_, nullptr, vTbuf);
        attn_kernel<<<attn_blocks, 64, attn_lds, stream>>>(qbuf, kbuf, vTbuf, edge_types, sp_dist,
                                                           edge_emb, dist_emb, obuf);
        gemm_wmma<BT_, D_, D_, 0><<<gemm_dd_blocks, 256, 0, stream>>>(obuf, oWt + dd, o_b + l * D_, h, nullptr);
        layernorm<<<BT_ / 8, 256, 0, stream>>>(h, ln2_w + l * D_, ln2_b + l * D_, x);
        gemm_wmma<BT_, DFF_, D_, 2><<<gemm_ff1_blocks, 256, 0, stream>>>(x, f1Wt + df, f1_b + l * DFF_, nullptr, ubuf);
        gemm_wmma<BT_, D_, DFF_, 0><<<gemm_dd_blocks, 256, 0, stream>>>(ubuf, f2Wt + df, f2_b + l * D_, h, nullptr);
    }

    hipMemcpyAsync(d_out, h, (size_t)BT_ * D_ * sizeof(float),
                   hipMemcpyDeviceToDevice, stream);
}
